// Add_Attn_Layer_59055800320841
// MI455X (gfx1250) — matrix operand order repaired and checked by reading, not compiled, not run on hardware
//
#include <hip/hip_runtime.h>

// ---------------------------------------------------------------------------
// Additive pairwise span scoring on MI455X (gfx1250, wave32).
//   out[b,i,j,c] = sum_d tanh(s[b,c,i,d] + e[b,c,j,d]) * v[d]
// B=2 C=8 L=256 D=128.
//
// Bound analysis: 8 MB of traffic (~0.35us @ 23.3 TB/s) vs 134M tanh evals
// -> TRANS-pipe bound. Strategy:
//   * native v_tanh_f32 (gfx1250) for the nonlinearity
//   * the sum_d (.)*v[d] reduction is fed to the WMMA pipe as
//     V_WMMA_F32_16X16X32_F16 with A = v-chunk replicated across rows and
//     B[k, j] = tanh tile (ISA B-layout places row j wholly in lane j's
//     slots, so each lane writes its own tanh results directly into its
//     B fragment — no cross-lane traffic).
//   * f32 accumulation; tanh in [-1,1], |v|<=0.1 -> f16 inputs are safe.
// One wave handles one 16(i) x 16(j) tile of one (b,c) slice.
// ---------------------------------------------------------------------------

typedef __attribute__((ext_vector_type(16))) _Float16 v16h;
typedef __attribute__((ext_vector_type(8)))  float    v8f;
typedef __attribute__((ext_vector_type(4)))  float    f32x4;

__device__ __forceinline__ float fast_tanh(float x) {
#if __has_builtin(__builtin_amdgcn_tanhf)
    return __builtin_amdgcn_tanhf(x);            // v_tanh_f32 (TRANS pipe)
#else
    // tanh(x) = (e^{2x}-1)/(e^{2x}+1), e^{2x} = exp2(2*log2(e)*x)
    float e = __builtin_amdgcn_exp2f(x * 2.8853900817779268f);
    return (e - 1.0f) * __builtin_amdgcn_rcpf(e + 1.0f);
#endif
}

__global__ void __launch_bounds__(32)
span_score_wmma_kernel(const float* __restrict__ start_hidden,
                       const float* __restrict__ end_hidden,
                       const float* __restrict__ v,
                       float* __restrict__ out) {
    constexpr int C = 8, L = 256, D = 128;

    // Grid decomposition: blockIdx.x = ((b*C + c) * 16 + ti) * 16 + tj
    const int blk = blockIdx.x;
    const int tj  = blk & 15;          // j-tile
    const int ti  = (blk >> 4) & 15;   // i-tile
    const int bc  = blk >> 8;          // b*C + c
    const int b   = bc >> 3;
    const int c   = bc & 7;

    const int lane = threadIdx.x;      // 0..31 (wave32)
    const int h    = lane >> 4;        // lane half: selects K sub-range
    const int jloc = lane & 15;        // this lane's column (j within tile)
    const int j    = tj * 16 + jloc;

    // ---- e_j resident in registers for the whole i-loop ------------------
    // B fragment, k order chosen here (A matches): lane (h*16+jloc) slot t holds
    // k = h*16 + t  ->  d = chunk*32 + h*16 + t. 64 floats/lane.
    const float* eptr = end_hidden + ((size_t)(bc * L + j)) * D;
    f32x4 er[4][4];
#pragma unroll
    for (int chunk = 0; chunk < 4; ++chunk) {
        const int d0 = chunk * 32 + h * 16;
#pragma unroll
        for (int q = 0; q < 4; ++q)
            er[chunk][q] = *(const f32x4*)(eptr + d0 + 4 * q);
    }

    // ---- A fragments: v-chunk replicated across all 16 rows --------------
    // A fragment, same k order as B here: slot t -> k = h*16 + t.
    v16h afrag[4];
#pragma unroll
    for (int chunk = 0; chunk < 4; ++chunk) {
        const float* vb = v + chunk * 32;
        f32x4 lo0 = *(const f32x4*)(vb + h * 16);
        f32x4 lo1 = *(const f32x4*)(vb + h * 16 + 4);
        f32x4 hi0 = *(const f32x4*)(vb + h * 16 + 8);
        f32x4 hi1 = *(const f32x4*)(vb + h * 16 + 12);
#pragma unroll
        for (int p = 0; p < 4; ++p) {
            afrag[chunk][p]      = (_Float16)lo0[p];
            afrag[chunk][4 + p]  = (_Float16)lo1[p];
            afrag[chunk][8 + p]  = (_Float16)hi0[p];
            afrag[chunk][12 + p] = (_Float16)hi1[p];
        }
    }

    // ---- loop over the 16 i's of this tile -------------------------------
#pragma unroll 1
    for (int i = 0; i < 16; ++i) {
        const int gi = ti * 16 + i;
        const float* sptr = start_hidden + ((size_t)(bc * L + gi)) * D;

        v8f acc = {0.f, 0.f, 0.f, 0.f, 0.f, 0.f, 0.f, 0.f};

#pragma unroll
        for (int chunk = 0; chunk < 4; ++chunk) {
            const int d0 = chunk * 32 + h * 16;
            // broadcast s_i slice (same address across a lane-half -> one
            // cacheline fetch), 64B as 4 x global_load_b128
            f32x4 s0 = *(const f32x4*)(sptr + d0);
            f32x4 s1 = *(const f32x4*)(sptr + d0 + 4);
            f32x4 s2 = *(const f32x4*)(sptr + d0 + 8);
            f32x4 s3 = *(const f32x4*)(sptr + d0 + 12);

            v16h bfrag;
#pragma unroll
            for (int p = 0; p < 4; ++p) {
                bfrag[p]      = (_Float16)fast_tanh(s0[p] + er[chunk][0][p]);
                bfrag[4 + p]  = (_Float16)fast_tanh(s1[p] + er[chunk][1][p]);
                bfrag[8 + p]  = (_Float16)fast_tanh(s2[p] + er[chunk][2][p]);
                bfrag[12 + p] = (_Float16)fast_tanh(s3[p] + er[chunk][3][p]);
            }

            // D[m, jloc] accumulates score_j (identical across rows m)
            acc = __builtin_amdgcn_wmma_f32_16x16x32_f16(
                false, afrag[chunk], false, bfrag,
                (short)0, acc, false, false);
        }

        // D layout: lane holds N = jloc; every row equals score_j, so
        // acc[0] is this lane's score. Halves are duplicates -> store once.
        const float score = acc[0];
        if (lane < 16) {
            out[((size_t)(b * L + gi) * L + j) * C + c] = score;
        }
    }
}

extern "C" void kernel_launch(void* const* d_in, const int* in_sizes, int n_in,
                              void* d_out, int out_size, void* d_ws, size_t ws_size,
                              hipStream_t stream) {
    (void)in_sizes; (void)n_in; (void)d_ws; (void)ws_size; (void)out_size;
    const float* start_hidden = (const float*)d_in[0];
    const float* end_hidden   = (const float*)d_in[1];
    const float* v            = (const float*)d_in[2];
    float* out                = (float*)d_out;

    constexpr int B = 2, C = 8, L = 256;
    const int blocks = B * C * (L / 16) * (L / 16);   // 4096 tiles
    span_score_wmma_kernel<<<blocks, 32, 0, stream>>>(start_hidden, end_hidden, v, out);
}
